// RecurrentBlock_35442070127168
// MI455X (gfx1250) — matrix operand order repaired and checked by reading, not compiled, not run on hardware
//
#include <hip/hip_runtime.h>

typedef unsigned short u16;
typedef __attribute__((ext_vector_type(16))) __bf16 v16bf;
typedef __attribute__((ext_vector_type(8)))  float  v8f;

#if defined(__has_builtin)
#  if __has_builtin(__builtin_amdgcn_global_load_async_to_lds_b128) && \
      __has_builtin(__builtin_amdgcn_s_wait_asynccnt)
#    define USE_ASYNC 1
#  endif
#endif
#ifndef USE_ASYNC
#  define USE_ASYNC 0
#endif

// exact parameter types of the async-load builtin (per hipcc diagnostics):
//   (v4i __device__* src_global, v4i as3* dst_lds, imm offset, imm cpol)
typedef int v4i __attribute__((vector_size(16)));
typedef __attribute__((address_space(1))) v4i* gptr_v4i;
typedef __attribute__((address_space(3))) v4i* lptr_v4i;
#define AS1(p) ((gptr_v4i)(uintptr_t)(p))
#define AS3(p) ((lptr_v4i)(unsigned)(uintptr_t)(p))

static __device__ __forceinline__ u16 f32_to_bf16(float f) {
    union { float f; unsigned u; } c; c.f = f;
    unsigned u = c.u;
    unsigned r = u + 0x7FFFu + ((u >> 16) & 1u);   // round-to-nearest-even
    return (u16)(r >> 16);
}

// ---------------------------------------------------------------------------
// Elementwise helpers
// ---------------------------------------------------------------------------
__global__ void cvt_f32_bf16(const float* __restrict__ in, u16* __restrict__ out, size_t n) {
    size_t i = (size_t)blockIdx.x * blockDim.x + threadIdx.x;
    size_t s = (size_t)gridDim.x * blockDim.x;
    for (; i < n; i += s) out[i] = f32_to_bf16(in[i]);
}

// conv_w [R_out, R_in, 4] f32  ->  4 matrices [R_out, R_in] bf16
__global__ void repack_conv(const float* __restrict__ cw, u16* __restrict__ out, size_t rr) {
    size_t n = rr * 4;
    size_t i = (size_t)blockIdx.x * blockDim.x + threadIdx.x;
    size_t s = (size_t)gridDim.x * blockDim.x;
    for (; i < n; i += s) {
        size_t k = i & 3, flat = i >> 2;           // (ro*R+ri, k)
        out[k * rr + flat] = f32_to_bf16(cw[i]);
    }
}

// rbuf holds r=sigmoid(..) -> becomes a; ibuf holds i=sigmoid(..) -> becomes b
__global__ void gate_fuse(float* __restrict__ rbuf, float* __restrict__ ibuf,
                          const float* __restrict__ hc, const float* __restrict__ lam,
                          int R, size_t n) {
    size_t i = (size_t)blockIdx.x * blockDim.x + threadIdx.x;
    size_t s = (size_t)gridDim.x * blockDim.x;
    for (; i < n; i += s) {
        int d = (int)(i % (size_t)R);
        float sp = log1pf(__expf(lam[d]));         // softplus
        float la = -8.0f * rbuf[i] * sp;
        float a  = __expf(la);
        float e2 = __expf(2.0f * la);
        float mult = sqrtf(fmaxf(1.0f - e2, 0.0f));
        rbuf[i] = a;
        ibuf[i] = mult * ibuf[i] * hc[i];
    }
}

// inclusive scan h_t = a_t * h_{t-1} + b_t over L; one thread per (b, r) channel
__global__ void rglru_scan(const float* __restrict__ a, const float* __restrict__ b,
                           u16* __restrict__ out, int Bn, int L, int R) {
    int idx = blockIdx.x * blockDim.x + threadIdx.x;
    if (idx >= Bn * R) return;
    int bb = idx / R, r = idx - bb * R;
    size_t base = (size_t)bb * L * R + r;
    float h = 0.0f;
    for (int t = 0; t < L; ++t) {
        size_t p = base + (size_t)t * R;
        h = fmaf(a[p], h, b[p]);
        out[p] = f32_to_bf16(h);
    }
}

// ---------------------------------------------------------------------------
// Generic bf16 WMMA GEMM:  out = epi( A_shifted * W^T + bias? + Cprev? )
//   A: [M,K] bf16 row-major (rows optionally time-shifted within batches of Lseq)
//   W: [N,K] bf16 row-major (so GEMM is act @ W^T)
// Block 256 threads = 8 waves (4 M-waves x 2 N-waves); tile 128x128x32;
// each wave computes 2x4 fragments of 16x16 via v_wmma_f32_16x16x32_bf16.
// Tiles staged in LDS via GLOBAL_LOAD_ASYNC_TO_LDS_B128 when available.
// ---------------------------------------------------------------------------
#define BM 128
#define BN 128
#define BK 32
#define LDR 40   // LDS row stride in elements (80B): conflict-free b128 reads

__global__ __launch_bounds__(256)
void gemm_bf16_wmma(const u16* __restrict__ A, int lda,
                    const u16* __restrict__ W, int ldb,
                    const float* __restrict__ bias,
                    const float* __restrict__ Cprev,
                    float* __restrict__ outF,
                    u16*   __restrict__ outB,
                    int M, int N, int K, int Lseq, int shift, int epi) {
    __shared__ __attribute__((aligned(16))) u16 As[2][BM * LDR];
    __shared__ __attribute__((aligned(16))) u16 Bs[2][BN * LDR];

    const int tid  = threadIdx.x;
    const int lane = tid & 31;
    const int wave = tid >> 5;
    const int wm   = wave & 3;    // 0..3  (M direction, 32 rows each)
    const int wn   = wave >> 2;   // 0..1  (N direction, 64 cols each)
    const int hlf  = lane >> 4;
    const int l16  = lane & 15;

    const int bm = blockIdx.y * BM;
    const int bn = blockIdx.x * BN;

    // global->LDS assignments: 128 rows, 2 threads/row, 16 elems (32B) each
    const int t_row = tid >> 1;
    const int t_col = (tid & 1) * 16;

    // time-shifted A source row (zero-fill outside the batch's sequence)
    const int gm = bm + t_row;
    const int bb = gm / Lseq;
    const int ls = (gm - bb * Lseq) + shift;
    const bool a_valid = ((unsigned)ls < (unsigned)Lseq);
    const size_t a_base = (size_t)(bb * Lseq + ls) * lda + t_col;
    const size_t b_base = (size_t)(bn + t_row) * ldb + t_col;

    const int lofs = t_row * LDR + t_col;

#if USE_ASYNC
    // zero-fill permanently-invalid (shifted OOB) A rows once, in both buffers
    if (!a_valid) {
        const uint4 z = make_uint4(0u, 0u, 0u, 0u);
        *(uint4*)(&As[0][lofs])     = z;
        *(uint4*)(&As[0][lofs + 8]) = z;
        *(uint4*)(&As[1][lofs])     = z;
        *(uint4*)(&As[1][lofs + 8]) = z;
    }
    auto issueAsync = [&](int k0, int buf) {
        if (a_valid) {
            __builtin_amdgcn_global_load_async_to_lds_b128(
                AS1(A + a_base + k0),     AS3(&As[buf][lofs]),     0, 0);
            __builtin_amdgcn_global_load_async_to_lds_b128(
                AS1(A + a_base + k0 + 8), AS3(&As[buf][lofs + 8]), 0, 0);
        }
        __builtin_amdgcn_global_load_async_to_lds_b128(
            AS1(W + b_base + k0),     AS3(&Bs[buf][lofs]),     0, 0);
        __builtin_amdgcn_global_load_async_to_lds_b128(
            AS1(W + b_base + k0 + 8), AS3(&Bs[buf][lofs + 8]), 0, 0);
    };
#else
    uint4 ra0, ra1, rb0, rb1;
    auto loadG = [&](int k0) {
        if (a_valid) {
            const uint4* pa = (const uint4*)(A + a_base + k0);
            ra0 = pa[0]; ra1 = pa[1];
        } else {
            ra0 = make_uint4(0u, 0u, 0u, 0u); ra1 = ra0;
        }
        const uint4* pb = (const uint4*)(W + b_base + k0);
        rb0 = pb[0]; rb1 = pb[1];
    };
    auto storeL = [&](int buf) {
        *(uint4*)(&As[buf][lofs])     = ra0;
        *(uint4*)(&As[buf][lofs + 8]) = ra1;
        *(uint4*)(&Bs[buf][lofs])     = rb0;
        *(uint4*)(&Bs[buf][lofs + 8]) = rb1;
    };
#endif

    v8f acc[2][4];
    const v8f vzero = {0.f, 0.f, 0.f, 0.f, 0.f, 0.f, 0.f, 0.f};
#pragma unroll
    for (int fm = 0; fm < 2; ++fm)
#pragma unroll
        for (int fn = 0; fn < 4; ++fn) acc[fm][fn] = vzero;

    union FAB { uint4 u[2]; v16bf v; };

    const int nk = K / BK;

#if USE_ASYNC
    issueAsync(0, 0);
    __builtin_amdgcn_s_wait_asynccnt(0);
    __syncthreads();
#else
    loadG(0);
    storeL(0);
    __syncthreads();
#endif

    for (int it = 0; it < nk; ++it) {
        const int cur = it & 1;
#if USE_ASYNC
        if (it + 1 < nk) issueAsync((it + 1) * BK, cur ^ 1);
#else
        if (it + 1 < nk) loadG((it + 1) * BK);
        if (it + 2 < nk) {                      // global_prefetch_b8 (CDNA5)
            __builtin_prefetch(A + a_base + (it + 2) * BK, 0, 3);
            __builtin_prefetch(W + b_base + (it + 2) * BK, 0, 3);
        }
#endif

        FAB af[2], bw[4];
#pragma unroll
        for (int fm = 0; fm < 2; ++fm) {
            const u16* p = &As[cur][(wm * 32 + fm * 16 + l16) * LDR];
            af[fm].u[0] = *(const uint4*)(p + hlf * 8);       // K = h*8 .. h*8+7
            af[fm].u[1] = *(const uint4*)(p + 16 + hlf * 8);  // K = 16+h*8 ..
        }
#pragma unroll
        for (int fn = 0; fn < 4; ++fn) {
            const u16* p = &Bs[cur][(wn * 64 + fn * 16 + l16) * LDR + hlf * 8];
            bw[fn].u[0] = *(const uint4*)(p);                 // K = h*8 .. h*8+7
            bw[fn].u[1] = *(const uint4*)(p + 16);            // K = 16+h*8 .. +7
        }
#pragma unroll
        for (int fm = 0; fm < 2; ++fm)
#pragma unroll
            for (int fn = 0; fn < 4; ++fn)
                acc[fm][fn] = __builtin_amdgcn_wmma_f32_16x16x32_bf16(
                    false, af[fm].v, false, bw[fn].v,
                    (short)0, acc[fm][fn], false, false);

#if USE_ASYNC
        if (it + 1 < nk) {
            __builtin_amdgcn_s_wait_asynccnt(0);
            __syncthreads();
        }
#else
        if (it + 1 < nk) {
            __syncthreads();
            storeL((it + 1) & 1);
            __syncthreads();
        }
#endif
    }

    // epilogue: C frag lane mapping -> row = hlf*8 + v, col = l16
#pragma unroll
    for (int fm = 0; fm < 2; ++fm) {
#pragma unroll
        for (int fn = 0; fn < 4; ++fn) {
            const int col = bn + wn * 64 + fn * 16 + l16;
            const float bv = bias ? bias[col] : 0.0f;
#pragma unroll
            for (int v = 0; v < 8; ++v) {
                const int row = bm + wm * 32 + fm * 16 + hlf * 8 + v;
                const size_t o = (size_t)row * N + col;
                float x = acc[fm][fn][v] + bv;
                if (Cprev) x += Cprev[o];
                if (epi == 1)      x = x / (1.0f + __expf(-x));    // SiLU
                else if (epi == 2) x = 1.0f / (1.0f + __expf(-x)); // sigmoid
                if (outF) outF[o] = x;
                if (outB) outB[o] = f32_to_bf16(x);
            }
        }
    }
}

// ---------------------------------------------------------------------------
extern "C" void kernel_launch(void* const* d_in, const int* in_sizes, int n_in,
                              void* d_out, int out_size, void* d_ws, size_t ws_size,
                              hipStream_t stream) {
    (void)in_sizes; (void)n_in; (void)out_size; (void)ws_size;
    const float* x   = (const float*)d_in[0];
    const float* w1  = (const float*)d_in[1];
    const float* b1  = (const float*)d_in[2];
    const float* cw  = (const float*)d_in[3];
    const float* cb  = (const float*)d_in[4];
    const float* wa  = (const float*)d_in[5];
    const float* ba  = (const float*)d_in[6];
    const float* wx  = (const float*)d_in[7];
    const float* bx  = (const float*)d_in[8];
    const float* lam = (const float*)d_in[9];
    const float* w2  = (const float*)d_in[10];
    const float* b2  = (const float*)d_in[11];
    float* out = (float*)d_out;

    const int Bn = 4, L = 2048, D = 2048, R = 2560;
    const int BL = Bn * L;                       // 8192
    const size_t nBLD = (size_t)BL * D;
    const size_t nBLR = (size_t)BL * R;
    const size_t nRD  = (size_t)R * D;
    const size_t nRR  = (size_t)R * R;

    char* p = (char*)d_ws;
    auto alloc = [&](size_t bytes) -> void* {
        void* q = (void*)p;
        p += (bytes + 255) & ~(size_t)255;
        return q;
    };
    u16*   xb   = (u16*)  alloc(nBLD * 2);   // x bf16
    u16*   w1b  = (u16*)  alloc(nRD  * 2);
    u16*   h1b  = (u16*)  alloc(nBLR * 2);   // silu(linear1) bf16
    u16*   cwb  = (u16*)  alloc(nRR * 4 * 2);// 4 x [R,R] bf16
    u16*   wab  = (u16*)  alloc(nRR  * 2);
    u16*   wxb  = (u16*)  alloc(nRR  * 2);
    u16*   w2b  = (u16*)  alloc(nRD  * 2);
    float* hcf  = (float*)alloc(nBLR * 4);   // conv out f32
    u16*   hcb  = (u16*)  alloc(nBLR * 2);   // conv out bf16
    float* af   = (float*)alloc(nBLR * 4);   // r -> a
    float* bf   = (float*)alloc(nBLR * 4);   // i -> b
    u16*   hsb  = (u16*)  alloc(nBLR * 2);   // scan output bf16

    const dim3 eb(256);
    const dim3 eg(2048);
    cvt_f32_bf16<<<eg, eb, 0, stream>>>(x,  xb,  nBLD);
    cvt_f32_bf16<<<eg, eb, 0, stream>>>(w1, w1b, nRD);
    repack_conv <<<eg, eb, 0, stream>>>(cw, cwb, nRR);
    cvt_f32_bf16<<<eg, eb, 0, stream>>>(wa, wab, nRR);
    cvt_f32_bf16<<<eg, eb, 0, stream>>>(wx, wxb, nRR);
    cvt_f32_bf16<<<eg, eb, 0, stream>>>(w2, w2b, nRD);

    auto gemm = [&](const u16* A, int lda, const u16* W, int ldb,
                    const float* bias, const float* Cp, float* oF, u16* oB,
                    int M, int N, int K, int Ls, int sh, int epi) {
        gemm_bf16_wmma<<<dim3(N / BN, M / BM), 256, 0, stream>>>(
            A, lda, W, ldb, bias, Cp, oF, oB, M, N, K, Ls, sh, epi);
    };

    // linear1 + SiLU : [BL,D] x [R,D]^T -> h1b bf16
    gemm(xb, D, w1b, D, b1, nullptr, nullptr, h1b, BL, R, D, BL, 0, /*silu*/1);

    // temporal conv: 4 shifted accumulated GEMM passes, offsets 2k-3
    for (int k = 0; k < 4; ++k) {
        gemm(h1b, R, cwb + (size_t)k * nRR, R,
             (k == 0) ? cb : nullptr,
             (k > 0) ? hcf : nullptr,
             hcf,
             (k == 3) ? hcb : nullptr,
             BL, R, R, L, 2 * k - 3, /*none*/0);
    }

    // gates: r = sigmoid(hc @ wa^T + ba), i = sigmoid(hc @ wx^T + bx)
    gemm(hcb, R, wab, R, ba, nullptr, af, nullptr, BL, R, R, BL, 0, /*sigmoid*/2);
    gemm(hcb, R, wxb, R, bx, nullptr, bf, nullptr, BL, R, R, BL, 0, /*sigmoid*/2);

    // a = exp(-8 r softplus(lam)); b = sqrt(1-a^2) * i * hc
    gate_fuse<<<eg, eb, 0, stream>>>(af, bf, hcf, lam, R, nBLR);

    // sequential RG-LRU scan over L (10240 channels)
    rglru_scan<<<dim3((Bn * R + 255) / 256), eb, 0, stream>>>(af, bf, hsb, Bn, L, R);

    // linear2: [BL,R] x [D,R]^T + b2 -> out f32
    gemm(hsb, R, w2b, R, b2, nullptr, out, nullptr, BL, D, R, BL, 0, /*none*/0);
}
